// UPFlowNetLite_4166118277409
// MI455X (gfx1250) — matrix operand order repaired and checked by reading, not compiled, not run on hardware
//
#include <hip/hip_runtime.h>
#include <hip/hip_bf16.h>
#include <math.h>

typedef _Float16 half_t;
typedef __attribute__((ext_vector_type(16))) _Float16 v16h;
typedef __attribute__((ext_vector_type(8)))  _Float16 v8h;
typedef __attribute__((ext_vector_type(8)))  float    v8f;
typedef __attribute__((ext_vector_type(4)))  unsigned int tdm_v4u;
typedef __attribute__((ext_vector_type(8)))  int          tdm_v8i;
typedef __attribute__((ext_vector_type(4)))  int          tdm_v4i;

#define DEVI __device__ __forceinline__

DEVI float lrelu_f(float v) { return v >= 0.f ? v : 0.1f * v; }
DEVI float act_f(float v, int act) {
  if (act == 1) return lrelu_f(v);
  if (act == 2) return fmaxf(v, 0.f);
  return v;
}

// ============ weight packing: OIHW f32 -> N-major [CoutP][KP] f16 ===========
__global__ void pack_weights_k(const float* __restrict__ w, half_t* __restrict__ wN,
                               int Cout, int Kreal, int KP, int CoutP) {
  long idx = (long)blockIdx.x * blockDim.x + threadIdx.x;
  long total = (long)CoutP * KP;
  if (idx >= total) return;
  int k = (int)(idx % KP);
  int n = (int)(idx / KP);
  half_t v = (half_t)0.f;
  if (n < Cout && k < Kreal) v = (half_t)w[(long)n * Kreal + k];
  wN[idx] = v;
}

// issue a 2-D TDM tile load: rows x KP f16 elements from gaddr -> LDS ldsAddr
DEVI void tdm_load_panel(unsigned ldsAddr, unsigned long long gaddr, int KP, int nrows) {
  tdm_v4u g0;
  g0[0] = 1u;                                              // count=1, valid user D#
  g0[1] = ldsAddr;                                         // lds_addr [63:32]
  g0[2] = (unsigned)(gaddr & 0xffffffffull);               // global_addr lo
  g0[3] = (unsigned)((gaddr >> 32) & 0x1ffffffull) | (2u << 30); // addr hi | type=2
  tdm_v8i g1;
  g1[0] = (int)(1u << 16);                                 // data_size = 2 bytes
  g1[1] = (int)(((unsigned)KP & 0xffffu) << 16);           // tensor_dim0 lo16
  g1[2] = (int)((((unsigned)KP >> 16) & 0xffffu) | ((unsigned)nrows << 16)); // dim0 hi | dim1 lo
  g1[3] = (int)(((unsigned)KP & 0xffffu) << 16);           // dim1 hi(0) | tile_dim0
  g1[4] = (int)(unsigned)nrows;                            // tile_dim1 | tile_dim2=0
  g1[5] = (int)(unsigned)KP;                               // tensor_dim0_stride lo32
  g1[6] = 0;                                               // stride hi | dim1_stride lo
  g1[7] = 0;
  tdm_v4i g2; g2[0] = 1; g2[1] = 1; g2[2] = 0; g2[3] = 0;  // tensor_dim2/3 = 1 (benign)
  tdm_v4i g3; g3[0] = 0; g3[1] = (int)(1u << 16); g3[2] = 0; g3[3] = 0; // tensor_dim4=1
#if __clang_major__ >= 23
  tdm_v8i gz8; gz8[0]=0;gz8[1]=0;gz8[2]=0;gz8[3]=0;gz8[4]=0;gz8[5]=0;gz8[6]=0;gz8[7]=0;
  __builtin_amdgcn_tensor_load_to_lds(g0, g1, g2, g3, gz8, 0);
#else
  __builtin_amdgcn_tensor_load_to_lds(g0, g1, g2, g3, 0);
#endif
}

// ===================== implicit-GEMM conv with WMMA =========================
// Wave tile: one output row, 32 consecutive x (2 x 16-pixel M tiles), NT cout
// tiles of 16. Wout % 32 == 0 and Hout % 4 == 0 for every layer here -> no
// divisions, no partial tiles, EXEC all-ones around WMMA.
// Weights: TDM-staged per block into LDS as [NT*16][KP] f16; B fragments are
// two ds b128 reads. A fragments are im2col gathers from global.
template <int KH, int KW, int NT>
__global__ void conv_wmma_k(const half_t* __restrict__ in, const half_t* __restrict__ wN,
                            const float* __restrict__ bias, half_t* __restrict__ out,
                            int Cin, int Hin, int Win,
                            int Cout, int Hout, int Wout,
                            int stride, int dil, int pad,
                            int Kreal, int KP, int nCT, int act) {
  extern __shared__ char smem_conv[];
  half_t* ldsW = (half_t*)smem_conv;

  const int lane = (int)(threadIdx.x & 31);
  const int wave = (int)(threadIdx.x >> 5);
  const int oy = (int)blockIdx.y * 4 + wave;
  const int xw = (int)blockIdx.x * 32;
  const int zb = (int)blockIdx.z;
  const int b  = zb / nCT;
  const int ct = zb - b * nCT;
  const int coutBase = ct * (NT * 16);

  // stage weight panel via Tensor Data Mover (wave 0 only), then block barrier
  if (wave == 0) {
    const unsigned long long gaddr = (unsigned long long)(const void*)(wN + (long)coutBase * KP);
    const unsigned ldsAddr = (unsigned)(unsigned long long)ldsW; // low 32 bits = LDS offset
    tdm_load_panel(ldsAddr, gaddr, KP, NT * 16);
    __builtin_amdgcn_s_wait_tensorcnt(0);
  }
  __syncthreads();

  const int am = lane & 15;
  const int hi = lane >> 4;
  const int akb = hi * 8;
  const int n = am;

  const half_t* inb = in + (long)b * Cin * Hin * Win;
  const int iy_base = oy * stride - pad;
  const int ixb0 = (xw + am) * stride - pad;
  const int ixb1 = (xw + 16 + am) * stride - pad;

  v8f acc[2][NT] = {};

  for (int k0 = 0; k0 < KP; k0 += 32) {
    v16h a[2];
#pragma unroll
    for (int mt = 0; mt < 2; ++mt) {
      const int ix_base = (mt == 0) ? ixb0 : ixb1;
#pragma unroll
      for (int j = 0; j < 16; ++j) {
        const int kk = k0 + ((j < 8) ? (akb + j) : (16 + akb + (j - 8)));
        half_t v = (half_t)0.f;
        if (kk < Kreal) {
          const int ci  = kk / (KH * KW);
          const int rem = kk - ci * (KH * KW);
          const int ky  = rem / KW;
          const int kx  = rem - ky * KW;
          const int iy = iy_base + ky * dil;
          const int ix = ix_base + kx * dil;
          if ((unsigned)iy < (unsigned)Hin && (unsigned)ix < (unsigned)Win)
            v = inb[((long)ci * Hin + iy) * Win + ix];
        }
        a[mt][j] = v;
      }
    }
#pragma unroll
    for (int nt = 0; nt < NT; ++nt) {
      const half_t* wrow = ldsW + (long)(nt * 16 + n) * KP + (k0 + hi * 8);
      v16h bm;
      *(v8h*)&bm       = *(const v8h*)(wrow);
      *((v8h*)&bm + 1) = *(const v8h*)(wrow + 16);
#pragma unroll
      for (int mt = 0; mt < 2; ++mt)
        acc[mt][nt] = __builtin_amdgcn_wmma_f32_16x16x32_f16(false, a[mt], false, bm,
                                                             (short)0, acc[mt][nt],
                                                             false, false);
    }
    if (k0 + 32 < KP) {
      const int ciN = (k0 + 32) / (KH * KW);
      const int iyc = iy_base < 0 ? 0 : iy_base;
      const int ixc = ixb0 < 0 ? 0 : ixb0;
      __builtin_prefetch(inb + ((long)ciN * Hin + iyc) * Win + ixc, 0, 1);
    }
  }

#pragma unroll
  for (int nt = 0; nt < NT; ++nt) {
    const int cout = coutBase + nt * 16 + n;
    if (cout < Cout) {
      const float bs = bias[cout];
      half_t* orow = out + (((long)b * Cout + cout) * Hout + oy) * Wout + xw;
#pragma unroll
      for (int mt = 0; mt < 2; ++mt) {
        v8h pk;
#pragma unroll
        for (int r = 0; r < 8; ++r)
          pk[r] = (half_t)act_f(acc[mt][nt][r] + bs, act);
        *(v8h*)(orow + mt * 16 + hi * 8) = pk;
      }
    }
  }
}

// ===================== elementwise / memory kernels =========================
__global__ void cvt_f32_to_f16_k(const float* __restrict__ in, half_t* __restrict__ out, long n) {
  long i = (long)blockIdx.x * blockDim.x + threadIdx.x;
  long st = (long)gridDim.x * blockDim.x;
  for (; i < n; i += st) out[i] = (half_t)in[i];
}

__global__ void copy_ch_h_k(const half_t* __restrict__ src, half_t* __restrict__ dst,
                            int B, int Cs, int Cd, int off, long HW) {
  long idx = (long)blockIdx.x * blockDim.x + threadIdx.x;
  long total = (long)B * Cs * HW;
  if (idx >= total) return;
  long b = idx / ((long)Cs * HW);
  long rem = idx - b * (long)Cs * HW;
  int c = (int)(rem / HW);
  long p = rem - (long)c * HW;
  dst[((long)b * Cd + off + c) * HW + p] = src[idx];
}

__global__ void copy_ch_f_k(const float* __restrict__ src, half_t* __restrict__ dst,
                            int B, int Cs, int Cd, int off, long HW) {
  long idx = (long)blockIdx.x * blockDim.x + threadIdx.x;
  long total = (long)B * Cs * HW;
  if (idx >= total) return;
  long b = idx / ((long)Cs * HW);
  long rem = idx - b * (long)Cs * HW;
  int c = (int)(rem / HW);
  long p = rem - (long)c * HW;
  dst[((long)b * Cd + off + c) * HW + p] = (half_t)src[idx];
}

// per-batch mean + inv-std over C*H*W (population var, eps=1e-16)
__global__ void meanvar_k(const half_t* __restrict__ x, float* __restrict__ mv, long nper) {
  const int b = blockIdx.x;
  const half_t* p = x + (long)b * nper;
  float s = 0.f, s2 = 0.f;
  for (long i = threadIdx.x; i < nper; i += blockDim.x) {
    float v = (float)p[i]; s += v; s2 += v * v;
  }
  __shared__ float sh[512];
  sh[threadIdx.x] = s; sh[threadIdx.x + 256] = s2;
  __syncthreads();
  for (int off = 128; off > 0; off >>= 1) {
    if ((int)threadIdx.x < off) {
      sh[threadIdx.x] += sh[threadIdx.x + off];
      sh[256 + threadIdx.x] += sh[256 + threadIdx.x + off];
    }
    __syncthreads();
  }
  if (threadIdx.x == 0) {
    float m = sh[0] / (float)nper;
    float var = sh[256] / (float)nper - m * m;
    mv[2 * b] = m;
    mv[2 * b + 1] = rsqrtf(var + 1e-16f);
  }
}

// 5x5 correlation of normalized features, fused lrelu, writes channels [0,25) of concat
__global__ void corr_lrelu_k(const half_t* __restrict__ x1, const half_t* __restrict__ x2,
                             const float* __restrict__ mv1, const float* __restrict__ mv2,
                             half_t* __restrict__ out, int B, int C, int H, int W, int Cd) {
  long HW = (long)H * W;
  long idx = (long)blockIdx.x * blockDim.x + threadIdx.x;
  long total = (long)B * HW;
  if (idx >= total) return;
  int b = (int)(idx / HW);
  long r = idx - (long)b * HW;
  int y = (int)(r / W), x = (int)(r - (long)y * W);
  float m1 = mv1[2 * b], is1 = mv1[2 * b + 1];
  float m2 = mv2[2 * b], is2 = mv2[2 * b + 1];
  const half_t* p1 = x1 + (long)b * C * HW;
  const half_t* p2 = x2 + (long)b * C * HW;
  float acc[25];
#pragma unroll
  for (int d = 0; d < 25; ++d) acc[d] = 0.f;
  for (int c = 0; c < C; ++c) {
    float a = ((float)p1[(long)c * HW + r] - m1) * is1;
    const half_t* row = p2 + (long)c * HW;
#pragma unroll
    for (int d = 0; d < 25; ++d) {
      int yy = y + d / 5 - 2, xx = x + d % 5 - 2;
      if ((unsigned)yy < (unsigned)H && (unsigned)xx < (unsigned)W)
        acc[d] += a * (((float)row[(long)yy * W + xx] - m2) * is2);
    }
  }
  const float invC = 1.f / (float)C;
#pragma unroll
  for (int d = 0; d < 25; ++d)
    out[((long)b * Cd + d) * HW + r] = (half_t)lrelu_f(acc[d] * invC);
}

// bilinear backwarp of f16 features with f32 flow (zero outside, per-tap masks)
__global__ void backwarp_h_k(const half_t* __restrict__ x, const float* __restrict__ flow,
                             half_t* __restrict__ out, int B, int C, int H, int W) {
  long HW = (long)H * W;
  long idx = (long)blockIdx.x * blockDim.x + threadIdx.x;
  long total = (long)B * HW;
  if (idx >= total) return;
  int b = (int)(idx / HW);
  long r = idx - (long)b * HW;
  int y = (int)(r / W), xq = (int)(r - (long)y * W);
  float px = (float)xq + flow[((long)b * 2 + 0) * HW + r];
  float py = (float)y + flow[((long)b * 2 + 1) * HW + r];
  float x0f = floorf(px), y0f = floorf(py);
  float wx = px - x0f, wy = py - y0f;
  int x0 = (int)x0f, y0 = (int)y0f;
  float w00 = (1.f - wx) * (1.f - wy), w10 = wx * (1.f - wy);
  float w01 = (1.f - wx) * wy, w11 = wx * wy;
  bool vx0 = (unsigned)x0 < (unsigned)W, vx1 = (unsigned)(x0 + 1) < (unsigned)W;
  bool vy0 = (unsigned)y0 < (unsigned)H, vy1 = (unsigned)(y0 + 1) < (unsigned)H;
  const half_t* p = x + (long)b * C * HW;
  for (int c = 0; c < C; ++c) {
    const half_t* pc = p + (long)c * HW;
    float v = 0.f;
    if (vx0 && vy0) v += w00 * (float)pc[(long)y0 * W + x0];
    if (vx1 && vy0) v += w10 * (float)pc[(long)y0 * W + x0 + 1];
    if (vx0 && vy1) v += w01 * (float)pc[(long)(y0 + 1) * W + x0];
    if (vx1 && vy1) v += w11 * (float)pc[(long)(y0 + 1) * W + x0 + 1];
    out[((long)b * C + c) * HW + r] = (half_t)v;
  }
}

// flow_out = backwarp(flow, inter_flow) * (1-sigmoid(mask)) + flow * sigmoid(mask)
__global__ void sgu_combine_k(const float* __restrict__ flow, const half_t* __restrict__ sgu,
                              float* __restrict__ out, int B, int H, int W) {
  long HW = (long)H * W;
  long idx = (long)blockIdx.x * blockDim.x + threadIdx.x;
  long total = (long)B * HW;
  if (idx >= total) return;
  int b = (int)(idx / HW);
  long r = idx - (long)b * HW;
  int y = (int)(r / W), xq = (int)(r - (long)y * W);
  float fx = (float)sgu[((long)b * 3 + 0) * HW + r];
  float fy = (float)sgu[((long)b * 3 + 1) * HW + r];
  float mk = (float)sgu[((long)b * 3 + 2) * HW + r];
  mk = 1.f / (1.f + expf(-mk));
  float px = (float)xq + fx, py = (float)y + fy;
  float x0f = floorf(px), y0f = floorf(py);
  float wx = px - x0f, wy = py - y0f;
  int x0 = (int)x0f, y0 = (int)y0f;
  float w00 = (1.f - wx) * (1.f - wy), w10 = wx * (1.f - wy);
  float w01 = (1.f - wx) * wy, w11 = wx * wy;
  bool vx0 = (unsigned)x0 < (unsigned)W, vx1 = (unsigned)(x0 + 1) < (unsigned)W;
  bool vy0 = (unsigned)y0 < (unsigned)H, vy1 = (unsigned)(y0 + 1) < (unsigned)H;
  for (int c = 0; c < 2; ++c) {
    const float* pc = flow + ((long)b * 2 + c) * HW;
    float v = 0.f;
    if (vx0 && vy0) v += w00 * pc[(long)y0 * W + x0];
    if (vx1 && vy0) v += w10 * pc[(long)y0 * W + x0 + 1];
    if (vx0 && vy1) v += w01 * pc[(long)(y0 + 1) * W + x0];
    if (vx1 && vy1) v += w11 * pc[(long)(y0 + 1) * W + x0 + 1];
    out[((long)b * 2 + c) * HW + r] = v * (1.f - mk) + pc[r] * mk;
  }
}

// bilinear 2x flow upsample (half-pixel centers, edge clamp) with magnitude rescale
__global__ void upflow_k(const float* __restrict__ in, float* __restrict__ out,
                         int B, int h0, int w0, int h, int w) {
  long HW = (long)h * w;
  long idx = (long)blockIdx.x * blockDim.x + threadIdx.x;
  long total = (long)B * 2 * HW;
  if (idx >= total) return;
  int bc = (int)(idx / HW);
  long r = idx - (long)bc * HW;
  int c = bc & 1, b = bc >> 1;
  int y = (int)(r / w), x = (int)(r - (long)y * w);
  float sy = (float)h0 / (float)h, sx = (float)w0 / (float)w;
  float py = ((float)y + 0.5f) * sy - 0.5f;
  float px = ((float)x + 0.5f) * sx - 0.5f;
  float y0f = floorf(py), x0f = floorf(px);
  float wy = py - y0f, wx = px - x0f;
  int y0 = (int)y0f, x0 = (int)x0f;
  int y0c = min(max(y0, 0), h0 - 1), y1c = min(max(y0 + 1, 0), h0 - 1);
  int x0c = min(max(x0, 0), w0 - 1), x1c = min(max(x0 + 1, 0), w0 - 1);
  const float* p = in + ((long)b * 2 + c) * (long)h0 * w0;
  float v = (1.f - wy) * ((1.f - wx) * p[(long)y0c * w0 + x0c] + wx * p[(long)y0c * w0 + x1c])
          + wy * ((1.f - wx) * p[(long)y1c * w0 + x0c] + wx * p[(long)y1c * w0 + x1c]);
  float scale = (c == 0) ? (float)w / (float)w0 : (float)h / (float)h0;
  out[idx] = v * scale;
}

__global__ void add_h_to_f_k(float* __restrict__ flow, const half_t* __restrict__ d, long n) {
  long i = (long)blockIdx.x * blockDim.x + threadIdx.x;
  if (i < n) flow[i] += (float)d[i];
}

// ===================== host orchestration ===================================
struct PW { half_t* wB; const float* bias; int Cout, Cin, KH, KW, Kreal, KP, CoutP; };

static inline unsigned gdim(long n) { return (unsigned)((n + 255) / 256); }

static void launch_conv(hipStream_t s, const PW& L, const half_t* in, half_t* out,
                        int B, int Hin, int Win, int stride, int dil, int act) {
  const int Hout = (Hin - 1) / stride + 1;
  const int Wout = (Win - 1) / stride + 1;
  const int NT = (L.CoutP >= 32) ? 2 : 1;
  const int nCT = L.CoutP / (16 * NT);
  dim3 grid((unsigned)(Wout / 32), (unsigned)(Hout / 4), (unsigned)(B * nCT));
  dim3 block(128, 1, 1);
  const int pad = dil * (L.KH / 2);
  const size_t shb = (size_t)NT * 16 * L.KP * sizeof(half_t);
  if (L.KH == 3) {
    if (NT == 2)
      conv_wmma_k<3, 3, 2><<<grid, block, shb, s>>>(in, L.wB, L.bias, out, L.Cin, Hin, Win,
                                                    L.Cout, Hout, Wout, stride, dil, pad,
                                                    L.Kreal, L.KP, nCT, act);
    else
      conv_wmma_k<3, 3, 1><<<grid, block, shb, s>>>(in, L.wB, L.bias, out, L.Cin, Hin, Win,
                                                    L.Cout, Hout, Wout, stride, dil, pad,
                                                    L.Kreal, L.KP, nCT, act);
  } else {
    if (NT == 2)
      conv_wmma_k<1, 1, 2><<<grid, block, shb, s>>>(in, L.wB, L.bias, out, L.Cin, Hin, Win,
                                                    L.Cout, Hout, Wout, stride, dil, pad,
                                                    L.Kreal, L.KP, nCT, act);
    else
      conv_wmma_k<1, 1, 1><<<grid, block, shb, s>>>(in, L.wB, L.bias, out, L.Cin, Hin, Win,
                                                    L.Cout, Hout, Wout, stride, dil, pad,
                                                    L.Kreal, L.KP, nCT, act);
  }
}

extern "C" void kernel_launch(void* const* d_in, const int* in_sizes, int n_in,
                              void* d_out, int out_size, void* d_ws, size_t ws_size,
                              hipStream_t stream) {
  (void)in_sizes; (void)out_size;
  if (n_in < 56) return;
  const int B = 4, H = 512, W = 1024;
  static const int ENC_CH[6] = {3, 16, 32, 64, 96, 128};

  // d_in flattening assumption (pytree order, dicts alphabetical):
  // 0:img1 1:img2 | enc 2..21 | est 22..29 | mid 30..37 | ref 38..49 | sgu 50..55
  struct WSpec { int wi, bi, Cout, Cin, KH, KW; };
  const WSpec specs[27] = {
    // encoder: stage s -> c1 (stride2), c2
    {2, 3, 16, 3, 3, 3},    {4, 5, 16, 16, 3, 3},
    {6, 7, 32, 16, 3, 3},   {8, 9, 32, 32, 3, 3},
    {10, 11, 64, 32, 3, 3}, {12, 13, 64, 64, 3, 3},
    {14, 15, 96, 64, 3, 3}, {16, 17, 96, 96, 3, 3},
    {18, 19, 128, 96, 3, 3},{20, 21, 128, 128, 3, 3},
    // est convs + last (indices 10..13)
    {22, 23, 64, 59, 3, 3}, {24, 25, 32, 64, 3, 3},
    {26, 27, 16, 32, 3, 3}, {28, 29, 2, 16, 1, 1},
    // mid 1x1 (indices 14..17)
    {30, 31, 32, 32, 1, 1}, {32, 33, 32, 64, 1, 1},
    {34, 35, 32, 96, 1, 1}, {36, 37, 32, 128, 1, 1},
    // ref convs + last (indices 18..23)
    {38, 39, 32, 18, 3, 3}, {40, 41, 32, 32, 3, 3}, {42, 43, 32, 32, 3, 3},
    {44, 45, 32, 32, 3, 3}, {46, 47, 32, 32, 3, 3}, {48, 49, 2, 32, 1, 1},
    // sgu (indices 24..26)
    {50, 51, 32, 64, 3, 3}, {52, 53, 16, 32, 3, 3}, {54, 55, 3, 16, 3, 3},
  };

  // ---- workspace bump allocator ----
  size_t used = 0;
  char* base = (char*)d_ws;
  auto alloc = [&](size_t bytes) -> char* {
    size_t off = (used + 255) & ~(size_t)255;
    used = off + bytes;
    return base + off;
  };

  PW pw[27];
  for (int i = 0; i < 27; ++i) {
    const WSpec& sp = specs[i];
    PW& L = pw[i];
    L.bias = (const float*)d_in[sp.bi];
    L.Cout = sp.Cout; L.Cin = sp.Cin; L.KH = sp.KH; L.KW = sp.KW;
    L.Kreal = sp.Cin * sp.KH * sp.KW;
    L.KP = (L.Kreal + 31) & ~31;
    L.CoutP = (L.Cout + 15) & ~15;
    L.wB = (half_t*)alloc((size_t)L.KP * L.CoutP * sizeof(half_t));
  }

  int Hl[6], Wl[6];
  for (int l = 0; l <= 5; ++l) { Hl[l] = H >> l; Wl[l] = W >> l; }
  half_t* imgh[2];
  half_t* pyr[2][6] = {};
  half_t* midb[2][6] = {};
  for (int i = 0; i < 2; ++i) imgh[i] = (half_t*)alloc((size_t)B * 3 * H * W * sizeof(half_t));
  for (int i = 0; i < 2; ++i)
    for (int l = 1; l <= 5; ++l)
      pyr[i][l] = (half_t*)alloc((size_t)B * ENC_CH[l] * Hl[l] * Wl[l] * sizeof(half_t));
  for (int i = 0; i < 2; ++i)
    for (int l = 2; l <= 5; ++l)
      midb[i][l] = (half_t*)alloc((size_t)B * 32 * Hl[l] * Wl[l] * sizeof(half_t));

  const size_t HW2 = (size_t)Hl[2] * Wl[2];
  half_t* warp    = (half_t*)alloc((size_t)B * 32 * HW2 * sizeof(half_t));
  half_t* concatb = (half_t*)alloc((size_t)B * 64 * HW2 * sizeof(half_t));
  half_t* tmpA    = (half_t*)alloc((size_t)B * 64 * HW2 * sizeof(half_t));
  half_t* tmpB    = (half_t*)alloc((size_t)B * 64 * HW2 * sizeof(half_t));
  half_t* sguOut  = (half_t*)alloc((size_t)B * 3 * HW2 * sizeof(half_t));
  half_t* delta   = (half_t*)alloc((size_t)B * 2 * HW2 * sizeof(half_t));
  float* flowA = (float*)alloc((size_t)B * 2 * HW2 * sizeof(float));
  float* flowB = (float*)alloc((size_t)B * 2 * HW2 * sizeof(float));
  float* mv1 = (float*)alloc(2 * B * sizeof(float));
  float* mv2 = (float*)alloc(2 * B * sizeof(float));
  if (used > ws_size) return;

  // ---- pack weights (f32 OIHW -> f16 N-major [Cout][K], zero padded) ----
  for (int i = 0; i < 27; ++i) {
    long total = (long)pw[i].KP * pw[i].CoutP;
    pack_weights_k<<<gdim(total), 256, 0, stream>>>((const float*)d_in[specs[i].wi], pw[i].wB,
                                                    pw[i].Cout, pw[i].Kreal, pw[i].KP, pw[i].CoutP);
  }

  // ---- images to f16 ----
  const long nimg = (long)B * 3 * H * W;
  cvt_f32_to_f16_k<<<gdim(nimg), 256, 0, stream>>>((const float*)d_in[0], imgh[0], nimg);
  cvt_f32_to_f16_k<<<gdim(nimg), 256, 0, stream>>>((const float*)d_in[1], imgh[1], nimg);

  // ---- encoder pyramids ----
  for (int i = 0; i < 2; ++i) {
    const half_t* cur = imgh[i];
    for (int s = 0; s < 5; ++s) {
      launch_conv(stream, pw[2 * s], cur, tmpA, B, Hl[s], Wl[s], 2, 1, 1);              // lrelu
      launch_conv(stream, pw[2 * s + 1], tmpA, pyr[i][s + 1], B, Hl[s + 1], Wl[s + 1], 1, 1, 2); // relu
      cur = pyr[i][s + 1];
    }
  }

  // ---- mid features (1x1, lrelu) ----
  for (int l = 2; l <= 5; ++l)
    for (int i = 0; i < 2; ++i)
      launch_conv(stream, pw[14 + (l - 2)], pyr[i][l], midb[i][l], B, Hl[l], Wl[l], 1, 1, 1);

  // ---- coarse-to-fine flow ----
  const long outOff[6] = {0, 0, 0, 262144, 327680, 344064};
  for (int l = 5; l >= 2; --l) {
    const int h = Hl[l], w = Wl[l];
    const long HWl = (long)h * w;
    const long npix = (long)B * HWl;
    const int C = ENC_CH[l];
    const half_t* x1 = pyr[0][l];
    const half_t* x2 = pyr[1][l];
    const half_t* x2w;
    if (l == 5) {
      hipMemsetAsync(flowA, 0, (size_t)B * 2 * HWl * sizeof(float), stream);
      x2w = x2;
    } else {
      upflow_k<<<gdim((long)B * 2 * HWl), 256, 0, stream>>>(flowA, flowB, B, Hl[l + 1], Wl[l + 1], h, w);
      // SGU: warp mid2, concat with mid1, 3 convs, blended warp of the flow
      backwarp_h_k<<<gdim(npix), 256, 0, stream>>>(midb[1][l], flowB, warp, B, 32, h, w);
      copy_ch_h_k<<<gdim((long)B * 32 * HWl), 256, 0, stream>>>(midb[0][l], concatb, B, 32, 64, 0, HWl);
      copy_ch_h_k<<<gdim((long)B * 32 * HWl), 256, 0, stream>>>(warp, concatb, B, 32, 64, 32, HWl);
      launch_conv(stream, pw[24], concatb, tmpA, B, h, w, 1, 1, 1);
      launch_conv(stream, pw[25], tmpA, tmpB, B, h, w, 1, 1, 1);
      launch_conv(stream, pw[26], tmpB, sguOut, B, h, w, 1, 1, 1);
      sgu_combine_k<<<gdim(npix), 256, 0, stream>>>(flowB, sguOut, flowA, B, h, w);
      backwarp_h_k<<<gdim(npix), 256, 0, stream>>>(x2, flowA, warp, B, C, h, w);
      x2w = warp;
    }
    // normalize + correlation, write concat [corr(25) | mid1(32) | flow(2)]
    meanvar_k<<<dim3(B), dim3(256), 0, stream>>>(x1, mv1, (long)C * HWl);
    meanvar_k<<<dim3(B), dim3(256), 0, stream>>>(x2w, mv2, (long)C * HWl);
    corr_lrelu_k<<<gdim(npix), 256, 0, stream>>>(x1, x2w, mv1, mv2, concatb, B, C, h, w, 59);
    copy_ch_h_k<<<gdim((long)B * 32 * HWl), 256, 0, stream>>>(midb[0][l], concatb, B, 32, 59, 25, HWl);
    copy_ch_f_k<<<gdim((long)B * 2 * HWl), 256, 0, stream>>>(flowA, concatb, B, 2, 59, 57, HWl);
    // estimator
    launch_conv(stream, pw[10], concatb, tmpA, B, h, w, 1, 1, 1);
    launch_conv(stream, pw[11], tmpA, tmpB, B, h, w, 1, 1, 1);
    launch_conv(stream, pw[12], tmpB, tmpA, B, h, w, 1, 1, 1);
    launch_conv(stream, pw[13], tmpA, delta, B, h, w, 1, 1, 0);
    add_h_to_f_k<<<gdim((long)B * 2 * HWl), 256, 0, stream>>>(flowA, delta, (long)B * 2 * HWl);
    // refinement (dilated), concat [x(16) | flow(2)]
    copy_ch_h_k<<<gdim((long)B * 16 * HWl), 256, 0, stream>>>(tmpA, concatb, B, 16, 18, 0, HWl);
    copy_ch_f_k<<<gdim((long)B * 2 * HWl), 256, 0, stream>>>(flowA, concatb, B, 2, 18, 16, HWl);
    launch_conv(stream, pw[18], concatb, tmpB, B, h, w, 1, 1, 1);
    launch_conv(stream, pw[19], tmpB, tmpA, B, h, w, 1, 2, 1);
    launch_conv(stream, pw[20], tmpA, tmpB, B, h, w, 1, 4, 1);
    launch_conv(stream, pw[21], tmpB, tmpA, B, h, w, 1, 8, 1);
    launch_conv(stream, pw[22], tmpA, tmpB, B, h, w, 1, 16, 1);
    launch_conv(stream, pw[23], tmpB, delta, B, h, w, 1, 1, 0);
    add_h_to_f_k<<<gdim((long)B * 2 * HWl), 256, 0, stream>>>(flowA, delta, (long)B * 2 * HWl);
    // emit this level's flow (output tuple order: level2, 3, 4, 5)
    hipMemcpyAsync((float*)d_out + outOff[l], flowA, (size_t)B * 2 * HWl * sizeof(float),
                   hipMemcpyDeviceToDevice, stream);
  }
}
